// PointNet2_76544907149654
// MI455X (gfx1250) — compile-verified, matrix operand order checked by reading, not run on hardware
//
#include <hip/hip_runtime.h>
#include <hip/hip_bf16.h>

// ---------------------------------------------------------------------------
// PointNet++ forward for MI455X (gfx1250). All matmuls use
// v_wmma_f32_16x16x32_f16 (f16 A/B, f32 accumulate). Weights are pre-packed
// per launch into f16 [Cout][Kp] (Kp = ceil32(Cin), zero padded) so every
// B-fragment is two contiguous 16B loads; activations are zero-padded to
// multiples of 32 channels so every A-fragment is two contiguous aligned
// vector loads — no per-element predication in the WMMA inner loops.
// GroupNorm (always groups of 8 channels) fuses into the WMMA epilogue via
// 8-lane __shfl_xor trees. PointConv MLPs run fully in LDS per center.
// ---------------------------------------------------------------------------

typedef __attribute__((ext_vector_type(16))) _Float16 v16h;
typedef __attribute__((ext_vector_type(8)))  _Float16 v8h;
typedef __attribute__((ext_vector_type(8)))  float    v8f;

#define BB      8
#define NN      4096
#define NOUT    4096
#define MM1     819
#define MM2     204
#define KNB     64
#define EPS_GN  1e-5f

// ---- fragment loaders (two contiguous aligned chunks per fragment) ---------
__device__ __forceinline__ v16h frag_from_h(const _Float16* p) {
  const v8h lo = *(const v8h*)(p);        // K = k0+base   .. +7
  const v8h hi = *(const v8h*)(p + 16);   // K = k0+base+16 .. +23
  return __builtin_shufflevector(lo, hi, 0,1,2,3,4,5,6,7,8,9,10,11,12,13,14,15);
}
__device__ __forceinline__ v16h frag_from_f(const float* p) {
  const v8f lo = *(const v8f*)(p);
  const v8f hi = *(const v8f*)(p + 16);
  v16h a;
  #pragma unroll
  for (int e = 0; e < 8; ++e) { a[e] = (_Float16)lo[e]; a[e + 8] = (_Float16)hi[e]; }
  return a;
}

// ------------------------------- FPS ---------------------------------------
__global__ void fps_kernel(const float* __restrict__ pos, int Np, int m,
                           int* __restrict__ idxOut, float* __restrict__ cposOut) {
  __shared__ float mind[4096];
  __shared__ float redV[256];
  __shared__ int   redI[256];
  __shared__ int   s_last;
  const int b = blockIdx.x;
  const float* p = pos + (size_t)b * Np * 3;
  for (int i = threadIdx.x; i < Np; i += blockDim.x) mind[i] = 1e10f;
  if (threadIdx.x == 0) s_last = 0;
  __syncthreads();
  for (int t = 0; t < m; ++t) {
    const int last = s_last;
    if (threadIdx.x == 0) {
      idxOut[(size_t)b * m + t] = last;
      cposOut[((size_t)b * m + t) * 3 + 0] = p[(size_t)last * 3 + 0];
      cposOut[((size_t)b * m + t) * 3 + 1] = p[(size_t)last * 3 + 1];
      cposOut[((size_t)b * m + t) * 3 + 2] = p[(size_t)last * 3 + 2];
    }
    const float lx = p[(size_t)last * 3 + 0];
    const float ly = p[(size_t)last * 3 + 1];
    const float lz = p[(size_t)last * 3 + 2];
    float bv = -1.0f; int bi = 0x7fffffff;
    for (int i = threadIdx.x; i < Np; i += blockDim.x) {
      const float ex = p[(size_t)i * 3 + 0] - lx;
      const float ey = p[(size_t)i * 3 + 1] - ly;
      const float ez = p[(size_t)i * 3 + 2] - lz;
      const float d  = ex * ex + ey * ey + ez * ez;
      const float mv = fminf(mind[i], d);
      mind[i] = mv;
      if (mv > bv) { bv = mv; bi = i; }
    }
    redV[threadIdx.x] = bv; redI[threadIdx.x] = bi;
    __syncthreads();
    for (int s = blockDim.x / 2; s > 0; s >>= 1) {
      if ((int)threadIdx.x < s) {
        const float ov = redV[threadIdx.x + s];
        const int   oi = redI[threadIdx.x + s];
        if (ov > redV[threadIdx.x] ||
            (ov == redV[threadIdx.x] && oi < redI[threadIdx.x])) {
          redV[threadIdx.x] = ov; redI[threadIdx.x] = oi;
        }
      }
      __syncthreads();
    }
    if (threadIdx.x == 0) s_last = redI[0];
    __syncthreads();
  }
}

// --------------------------- radius ball query ------------------------------
__global__ void ball_kernel(const float* __restrict__ pos,
                            const float* __restrict__ centers,
                            int Np, int M, float r2,
                            int* __restrict__ nbrOut) {
  __shared__ float d2s[4096];
  __shared__ float redV[256];
  __shared__ int   redI[256];
  __shared__ int   s_done;
  const int g = blockIdx.x;
  const int b = g / M, m = g % M;
  const float* p = pos + (size_t)b * Np * 3;
  const float cx = centers[((size_t)b * M + m) * 3 + 0];
  const float cy = centers[((size_t)b * M + m) * 3 + 1];
  const float cz = centers[((size_t)b * M + m) * 3 + 2];
  for (int i = threadIdx.x; i < Np; i += blockDim.x) {
    const float ex = p[(size_t)i * 3 + 0] - cx;
    const float ey = p[(size_t)i * 3 + 1] - cy;
    const float ez = p[(size_t)i * 3 + 2] - cz;
    d2s[i] = ex * ex + ey * ey + ez * ez;
  }
  if (threadIdx.x == 0) s_done = 0;
  __syncthreads();
  int* nb = nbrOut + (size_t)g * KNB;
  for (int s = 0; s < KNB; ++s) {
    float bv = 1e30f; int bi = 0x7fffffff;
    if (!s_done) {
      for (int i = threadIdx.x; i < Np; i += blockDim.x) {
        const float v = d2s[i];
        if (v < bv) { bv = v; bi = i; }
      }
    }
    redV[threadIdx.x] = bv; redI[threadIdx.x] = bi;
    __syncthreads();
    for (int st = blockDim.x / 2; st > 0; st >>= 1) {
      if ((int)threadIdx.x < st) {
        const float ov = redV[threadIdx.x + st];
        const int   oi = redI[threadIdx.x + st];
        if (ov < redV[threadIdx.x] ||
            (ov == redV[threadIdx.x] && oi < redI[threadIdx.x])) {
          redV[threadIdx.x] = ov; redI[threadIdx.x] = oi;
        }
      }
      __syncthreads();
    }
    if (threadIdx.x == 0) {
      if (s_done || redV[0] > r2) { nb[s] = -1; s_done = 1; }
      else { nb[s] = redI[0]; d2s[redI[0]] = 1e30f; }
    }
    __syncthreads();
  }
}

// --------------------------- weight pre-pack --------------------------------
// W [Cin, Cout] f32 (row-major)  ->  Wt [CoutPad][Kp] f16, zero padded.
__global__ void pack_w_kernel(const float* __restrict__ W, _Float16* __restrict__ Wt,
                              int Cin, int Cout, int Kp, int CoutPad) {
  const int t = blockIdx.x * blockDim.x + threadIdx.x;
  if (t >= Kp * CoutPad) return;
  const int col = t / Kp, k = t % Kp;
  Wt[t] = (col < Cout && k < Cin) ? (_Float16)W[(size_t)k * Cout + col]
                                  : (_Float16)0.0f;
}

// --------------------- GroupNorm epilogue (8-channel groups) ----------------
__device__ __forceinline__ float gn_epilogue(float v, float gamma, float beta) {
  float s = v, s2 = v * v;
  #pragma unroll
  for (int off = 1; off < 8; off <<= 1) {
    s  += __shfl_xor(s,  off, 32);
    s2 += __shfl_xor(s2, off, 32);
  }
  const float mu  = s * 0.125f;
  const float var = s2 * 0.125f - mu * mu;
  return (v - mu) * rsqrtf(var + EPS_GN) * gamma + beta;
}

// -------------------- fused PointConv (gather+MLP+max) ----------------------
#define PC_ROWS 64
#define PCA_W   160   // ceil32(131); rows zero-padded by the gather
#define PCB_W   256

template <int WIN, int WOUT>
__device__ __forceinline__ void pc_layer(const _Float16 (*in)[WIN],
                                         _Float16 (*outb)[WOUT],
                                         const _Float16* __restrict__ Wt, int Kp,
                                         const float* __restrict__ bias,
                                         const float* __restrict__ gamma,
                                         const float* __restrict__ beta,
                                         int Cout) {
  const int lane = threadIdx.x & 31;
  const int wave = threadIdx.x >> 5;           // row tile 0..3
  const int half = lane >> 4;
  const int l16  = lane & 15;
  const _Float16* arow = &in[wave * 16 + l16][0];
  for (int ct = 0; ct < Cout / 16; ++ct) {
    const int col = ct * 16 + l16;
    const _Float16* brow = Wt + (size_t)col * Kp;
    v8f acc = {};
    for (int k0 = 0; k0 < Kp; k0 += 32) {
      const v16h a  = frag_from_h(arow + k0 + half * 8);
      const v16h bf = frag_from_h(brow + k0 + half * 8);
      acc = __builtin_amdgcn_wmma_f32_16x16x32_f16(false, a, false, bf,
                                                   (short)0, acc, false, false);
    }
    const float bb = bias[col], gg = gamma[col], be = beta[col];
    #pragma unroll
    for (int r = 0; r < 8; ++r) {
      const int row = wave * 16 + r + half * 8;
      float v = fmaxf(acc[r] + bb, 0.0f);
      v = gn_epilogue(v, gg, be);
      outb[row][col] = (_Float16)v;
    }
  }
}

__global__ void point_conv_kernel(
    const float* __restrict__ feat, const float* __restrict__ pos,
    const float* __restrict__ centers, const int* __restrict__ nbr,
    const _Float16* Wt1, int Kp1, const float* b1, const float* ga1, const float* be1,
    const _Float16* Wt2, int Kp2, const float* b2, const float* ga2, const float* be2,
    const _Float16* Wt3, int Kp3, const float* b3, const float* ga3, const float* be3,
    float* __restrict__ out,
    int Np, int M, int Cf, int C1, int C2, int C3) {
  __shared__ _Float16 bufA[PC_ROWS][PCA_W];
  __shared__ _Float16 bufB[PC_ROWS][PCB_W];
  const int g = blockIdx.x;
  const int b = g / M, m = g % M;
  const float* fB = feat + (size_t)b * Np * Cf;
  const float* pB = pos  + (size_t)b * Np * 3;
  const float cx = centers[((size_t)b * M + m) * 3 + 0];
  const float cy = centers[((size_t)b * M + m) * 3 + 1];
  const float cz = centers[((size_t)b * M + m) * 3 + 2];
  const int* nb = nbr + ((size_t)b * M + m) * KNB;
  const int Cin = Cf + 3;
  // gather [K, Cf+3] = [x_j, pos_j - center]; zero-fill full padded width
  for (int t = threadIdx.x; t < PC_ROWS * PCA_W; t += blockDim.x) {
    const int j = t / PCA_W, c = t % PCA_W;
    float v = 0.0f;
    const int n = nb[j];
    if (n >= 0) {
      if (c < Cf)       v = fB[(size_t)n * Cf + c];
      else if (c < Cin) {
        const int d = c - Cf;
        v = pB[(size_t)n * 3 + d] - (d == 0 ? cx : (d == 1 ? cy : cz));
      }
    }
    bufA[j][c] = (_Float16)v;
  }
  __syncthreads();
  pc_layer<PCA_W, PCB_W>(bufA, bufB, Wt1, Kp1, b1, ga1, be1, C1);
  __syncthreads();
  pc_layer<PCB_W, PCA_W>(bufB, bufA, Wt2, Kp2, b2, ga2, be2, C2);
  __syncthreads();
  pc_layer<PCA_W, PCB_W>(bufA, bufB, Wt3, Kp3, b3, ga3, be3, C3);
  __syncthreads();
  for (int c = threadIdx.x; c < C3; c += blockDim.x) {
    float mx = -1e10f;
    for (int j = 0; j < KNB; ++j)
      if (nb[j] >= 0) mx = fmaxf(mx, (float)bufB[j][c]);
    out[((size_t)b * M + m) * C3 + c] = mx;
  }
}

// ----------------------- generic WMMA linear kernel -------------------------
// X: [R, ldin] f32, zero-padded so columns [Cin, Kp) are readable/zero.
// Wt: packed [CoutPad][Kp] f16. ldin and Kp are multiples of 32.
__global__ void linear_wmma_kernel(const float* __restrict__ X,
                                   const _Float16* __restrict__ Wt,
                                   const float* __restrict__ bias,
                                   const float* __restrict__ gamma,
                                   const float* __restrict__ beta,
                                   float* __restrict__ Y,
                                   int R, int Kp, int Cout, int ldin,
                                   int do_relu, int do_gn) {
  const int lane = threadIdx.x & 31;
  const int wave = threadIdx.x >> 5;                 // 8 waves / block
  const int rowBase = (blockIdx.x * 8 + wave) * 16;
  if (rowBase >= R) return;                          // wave-uniform exit
  const int half = lane >> 4;
  const int l16  = lane & 15;
  const int row  = rowBase + l16;
  const int rowc = (row < R) ? row : (R - 1);        // clamp, discard at store
  const int col  = blockIdx.y * 16 + l16;
  const float*    arow = X  + (size_t)rowc * ldin + half * 8;
  const _Float16* brow = Wt + (size_t)col  * Kp   + half * 8;
  v8f acc = {};
  for (int k0 = 0; k0 < Kp; k0 += 32) {
    const v16h a  = frag_from_f(arow + k0);
    const v16h bf = frag_from_h(brow + k0);
    acc = __builtin_amdgcn_wmma_f32_16x16x32_f16(false, a, false, bf,
                                                 (short)0, acc, false, false);
  }
  const float bb = (col < Cout) ? bias[col] : 0.0f;
  const float gg = (do_gn && col < Cout) ? gamma[col] : 1.0f;
  const float be = (do_gn && col < Cout) ? beta[col]  : 0.0f;
  #pragma unroll
  for (int r = 0; r < 8; ++r) {
    const int orow = rowBase + r + half * 8;
    float v = acc[r] + bb;
    if (do_relu) v = fmaxf(v, 0.0f);
    if (do_gn)   v = gn_epilogue(v, gg, be);
    if (orow < R && col < Cout) Y[(size_t)orow * Cout + col] = v;
  }
}

// ------------------------------ small helpers -------------------------------
__global__ void colmax_kernel(const float* __restrict__ in, float* __restrict__ out,
                              int Rows, int C) {
  const int b = blockIdx.x;
  for (int c = threadIdx.x; c < C; c += blockDim.x) {
    float mx = -1e30f;
    for (int r = 0; r < Rows; ++r)
      mx = fmaxf(mx, in[((size_t)b * Rows + r) * C + c]);
    out[(size_t)b * C + c] = mx;
  }
}

// dst [rows, dstW] with zero pad columns beyond wA+wB
__global__ void concat2_kernel(const float* __restrict__ A, int wA,
                               const float* __restrict__ Bs, int wB,
                               float* __restrict__ dst, long total, int dstW) {
  const long t = (long)blockIdx.x * blockDim.x + threadIdx.x;
  if (t >= total) return;
  const long r = t / dstW; const int c = (int)(t % dstW);
  float v = 0.0f;
  if (c < wA)            v = A[r * wA + c];
  else if (c < wA + wB)  v = Bs[r * wB + (c - wA)];
  dst[t] = v;
}

__global__ void concat_bcast_kernel(const float* __restrict__ gvec, int wG,
                                    const float* __restrict__ Bs, int wB,
                                    float* __restrict__ dst, long total,
                                    int rowsPerBatch) {
  const long t = (long)blockIdx.x * blockDim.x + threadIdx.x;
  const int w = wG + wB;
  if (t >= total) return;
  const long r = t / w; const int c = (int)(t % w);
  const long b = r / rowsPerBatch;
  dst[t] = (c < wG) ? gvec[b * wG + c] : Bs[r * wB + (c - wG)];
}

__global__ void knn3_interp_kernel(const float* __restrict__ src,   // [B,Ns,C]
                                   const float* __restrict__ spos,  // [B,Ns,3]
                                   const float* __restrict__ dpos,  // [B,Nd,3]
                                   float* __restrict__ out,         // [B,Nd,C]
                                   int Ns, int Nd, int C, int total) {
  const int gid = blockIdx.x * blockDim.x + threadIdx.x;
  if (gid >= total) return;
  const int b = gid / Nd;
  const float* sp = spos + (size_t)b * Ns * 3;
  const float dx = dpos[(size_t)gid * 3 + 0];
  const float dy = dpos[(size_t)gid * 3 + 1];
  const float dz = dpos[(size_t)gid * 3 + 2];
  float bd0 = 1e30f, bd1 = 1e30f, bd2 = 1e30f;
  int   bi0 = 0,     bi1 = 0,     bi2 = 0;
  for (int i = 0; i < Ns; ++i) {
    const float ex = sp[(size_t)i * 3 + 0] - dx;
    const float ey = sp[(size_t)i * 3 + 1] - dy;
    const float ez = sp[(size_t)i * 3 + 2] - dz;
    const float d2 = ex * ex + ey * ey + ez * ez;
    if (d2 < bd0)      { bd2 = bd1; bi2 = bi1; bd1 = bd0; bi1 = bi0; bd0 = d2; bi0 = i; }
    else if (d2 < bd1) { bd2 = bd1; bi2 = bi1; bd1 = d2;  bi1 = i; }
    else if (d2 < bd2) { bd2 = d2;  bi2 = i; }
  }
  float w0 = 1.0f / fmaxf(bd0, 1e-16f);
  float w1 = 1.0f / fmaxf(bd1, 1e-16f);
  float w2 = 1.0f / fmaxf(bd2, 1e-16f);
  const float wsum = w0 + w1 + w2;
  w0 /= wsum; w1 /= wsum; w2 /= wsum;
  const float* s0 = src + ((size_t)b * Ns + bi0) * C;
  const float* s1 = src + ((size_t)b * Ns + bi1) * C;
  const float* s2 = src + ((size_t)b * Ns + bi2) * C;
  for (int c = 0; c < C; ++c)
    out[(size_t)gid * C + c] = w0 * s0[c] + w1 * s1[c] + w2 * s2[c];
}

__global__ void logsoftmax_kernel(const float* __restrict__ in,
                                  float* __restrict__ out, int rows, int C) {
  const int r = blockIdx.x * blockDim.x + threadIdx.x;
  if (r >= rows) return;
  float m = -1e30f;
  for (int c = 0; c < C; ++c) m = fmaxf(m, in[(size_t)r * C + c]);
  float s = 0.0f;
  for (int c = 0; c < C; ++c) s += expf(in[(size_t)r * C + c] - m);
  const float lse = m + logf(s);
  for (int c = 0; c < C; ++c)
    out[(size_t)r * C + c] = in[(size_t)r * C + c] - lse;
}

// ------------------------------- host driver --------------------------------
// Param flattening assumption (JAX sorted-key dict flatten):
//   x, input_pts, output_pts, then params:
//   fp1(3 layers), fp2(2), fp3(2), lin1{W,b}, lin2{W,b}, lin3{W,b},
//   sa1(3), sa2(3), sa3(3); each MLP layer flattens as W, b, beta, gamma.
enum {
  IN_X = 0, IN_IPTS = 1, IN_OPTS = 2,
  FP1_L0 = 3, FP1_L1 = 7, FP1_L2 = 11,
  FP2_L0 = 15, FP2_L1 = 19,
  FP3_L0 = 23, FP3_L1 = 27,
  LIN1_W = 31, LIN2_W = 33, LIN3_W = 35,
  SA1_L0 = 37, SA1_L1 = 41, SA1_L2 = 45,
  SA2_L0 = 49, SA2_L1 = 53, SA2_L2 = 57,
  SA3_L0 = 61, SA3_L1 = 65, SA3_L2 = 69
};

static inline int cdiv(int a, int b) { return (a + b - 1) / b; }

struct PackedW { _Float16* p; int Kp; int CoutPad; };

extern "C" void kernel_launch(void* const* d_in, const int* in_sizes, int n_in,
                              void* d_out, int out_size, void* d_ws, size_t ws_size,
                              hipStream_t stream) {
  (void)in_sizes; (void)n_in; (void)out_size; (void)ws_size;
  const float* x       = (const float*)d_in[IN_X];
  const float* in_pts  = (const float*)d_in[IN_IPTS];
  const float* out_pts = (const float*)d_in[IN_OPTS];
  #define PW(i)    ((const float*)d_in[(i)])
  #define LB(b_)   PW((b_) + 1)
  #define LBE(b_)  PW((b_) + 2)
  #define LGA(b_)  PW((b_) + 3)

  // ---- workspace carve ----
  char* ws = (char*)d_ws;
  size_t off = 0;
  auto carve = [&](size_t bytes) -> void* {
    void* p = ws + off;
    off = (off + bytes + 255) & ~(size_t)255;
    return p;
  };
  int*   idx1  = (int*)  carve((size_t)BB * MM1 * 4);
  float* pos1  = (float*)carve((size_t)BB * MM1 * 3 * 4);
  int*   idx2  = (int*)  carve((size_t)BB * MM2 * 4);
  float* pos2  = (float*)carve((size_t)BB * MM2 * 3 * 4);
  int*   nbr1  = (int*)  carve((size_t)BB * MM1 * KNB * 4);
  int*   nbr2  = (int*)  carve((size_t)BB * MM2 * KNB * 4);
  float* x1    = (float*)carve((size_t)BB * MM1 * 128 * 4);
  float* x2    = (float*)carve((size_t)BB * MM2 * 256 * 4);
  float* gbuf  = (float*)carve((size_t)BB * 1024 * 4);
  const size_t arenaElems = (size_t)BB * NOUT * 128;
  float* arA   = (float*)carve(arenaElems * 4);
  float* arB   = (float*)carve(arenaElems * 4);

  // ---- pack all weights to f16 [CoutPad][Kp] ----
  auto pack = [&](int base, int Cin, int Cout) -> PackedW {
    const int Kp = (Cin + 31) & ~31;
    const int Cp = (Cout + 15) & ~15;
    _Float16* p = (_Float16*)carve((size_t)Kp * Cp * sizeof(_Float16));
    pack_w_kernel<<<cdiv(Kp * Cp, 256), 256, 0, stream>>>(PW(base), p, Cin, Cout, Kp, Cp);
    return {p, Kp, Cp};
  };
  PackedW sa1w0 = pack(SA1_L0,    6,   64);
  PackedW sa1w1 = pack(SA1_L1,   64,   64);
  PackedW sa1w2 = pack(SA1_L2,   64,  128);
  PackedW sa2w0 = pack(SA2_L0,  131,  128);
  PackedW sa2w1 = pack(SA2_L1,  128,  128);
  PackedW sa2w2 = pack(SA2_L2,  128,  256);
  PackedW sa3w0 = pack(SA3_L0,  259,  256);
  PackedW sa3w1 = pack(SA3_L1,  256,  512);
  PackedW sa3w2 = pack(SA3_L2,  512, 1024);
  PackedW fp3w0 = pack(FP3_L0, 1280,  256);
  PackedW fp3w1 = pack(FP3_L1,  256,  256);
  PackedW fp2w0 = pack(FP2_L0,  384,  256);
  PackedW fp2w1 = pack(FP2_L1,  256,  128);
  PackedW fp1w0 = pack(FP1_L0,  128,  128);
  PackedW fp1w1 = pack(FP1_L1,  128,  128);
  PackedW fp1w2 = pack(FP1_L2,  128,  128);
  PackedW lin1w = pack(LIN1_W,  128,  128);
  PackedW lin2w = pack(LIN2_W,  128,  128);
  PackedW lin3w = pack(LIN3_W,  128,   13);

  auto linear = [&](const float* X, const PackedW& w, int base, float* Y,
                    int R, int Cout, int ldin, int relu, int gn) {
    dim3 grid(cdiv(cdiv(R, 16), 8), w.CoutPad / 16);
    linear_wmma_kernel<<<grid, 256, 0, stream>>>(
        X, w.p, LB(base), gn ? LGA(base) : nullptr, gn ? LBE(base) : nullptr,
        Y, R, w.Kp, Cout, ldin, relu, gn);
  };

  // ---- SA1 ----
  fps_kernel<<<BB, 256, 0, stream>>>(in_pts, NN, MM1, idx1, pos1);
  ball_kernel<<<BB * MM1, 256, 0, stream>>>(in_pts, pos1, NN, MM1, 0.2f * 0.2f, nbr1);
  point_conv_kernel<<<BB * MM1, 128, 0, stream>>>(
      x, in_pts, pos1, nbr1,
      sa1w0.p, sa1w0.Kp, LB(SA1_L0), LGA(SA1_L0), LBE(SA1_L0),
      sa1w1.p, sa1w1.Kp, LB(SA1_L1), LGA(SA1_L1), LBE(SA1_L1),
      sa1w2.p, sa1w2.Kp, LB(SA1_L2), LGA(SA1_L2), LBE(SA1_L2),
      x1, NN, MM1, 3, 64, 64, 128);
  // ---- SA2 ----
  fps_kernel<<<BB, 256, 0, stream>>>(pos1, MM1, MM2, idx2, pos2);
  ball_kernel<<<BB * MM2, 256, 0, stream>>>(pos1, pos2, MM1, MM2, 0.4f * 0.4f, nbr2);
  point_conv_kernel<<<BB * MM2, 128, 0, stream>>>(
      x1, pos1, pos2, nbr2,
      sa2w0.p, sa2w0.Kp, LB(SA2_L0), LGA(SA2_L0), LBE(SA2_L0),
      sa2w1.p, sa2w1.Kp, LB(SA2_L1), LGA(SA2_L1), LBE(SA2_L1),
      sa2w2.p, sa2w2.Kp, LB(SA2_L2), LGA(SA2_L2), LBE(SA2_L2),
      x2, MM1, MM2, 128, 128, 128, 256);
  // ---- SA3 (global) ----
  {
    const int R = BB * MM2;
    const long tot = (long)R * 288;                  // ceil32(259), zero padded
    concat2_kernel<<<cdiv((int)tot, 256), 256, 0, stream>>>(x2, 256, pos2, 3, arA, tot, 288);
    linear(arA, sa3w0, SA3_L0, arB, R,  256, 288, 1, 1);
    linear(arB, sa3w1, SA3_L1, arA, R,  512, 256, 1, 1);
    linear(arA, sa3w2, SA3_L2, arB, R, 1024, 512, 1, 1);
    colmax_kernel<<<BB, 256, 0, stream>>>(arB, gbuf, MM2, 1024);
  }
  // ---- FP3 ----
  {
    const int R = BB * MM2;
    const long tot = (long)R * 1280;
    concat_bcast_kernel<<<cdiv((int)tot, 256), 256, 0, stream>>>(
        gbuf, 1024, x2, 256, arA, tot, MM2);
    linear(arA, fp3w0, FP3_L0, arB, R, 256, 1280, 1, 1);
    linear(arB, fp3w1, FP3_L1, arA, R, 256,  256, 1, 1);   // f3 in arA
  }
  // ---- FP2 ----
  {
    const int R = BB * MM1;
    knn3_interp_kernel<<<cdiv(R, 256), 256, 0, stream>>>(
        arA, pos2, pos1, arB, MM2, MM1, 256, R);
    const long tot = (long)R * 384;
    concat2_kernel<<<cdiv((int)tot, 256), 256, 0, stream>>>(arB, 256, x1, 128, arA, tot, 384);
    linear(arA, fp2w0, FP2_L0, arB, R, 256, 384, 1, 1);
    linear(arB, fp2w1, FP2_L1, arA, R, 128, 256, 1, 1);    // f2 in arA
  }
  // ---- FP1 + head ----
  {
    const int R = BB * NOUT;
    knn3_interp_kernel<<<cdiv(R, 256), 256, 0, stream>>>(
        arA, pos1, out_pts, arB, MM1, NOUT, 128, R);
    linear(arB, fp1w0, FP1_L0, arA, R, 128, 128, 1, 1);
    linear(arA, fp1w1, FP1_L1, arB, R, 128, 128, 1, 1);
    linear(arB, fp1w2, FP1_L2, arA, R, 128, 128, 1, 1);    // f1 in arA
    linear(arA, lin1w, LIN1_W, arB, R, 128, 128, 1, 0);    // relu, no GN
    linear(arB, lin2w, LIN2_W, arA, R, 128, 128, 0, 0);
    linear(arA, lin3w, LIN3_W, arB, R,  13, 128, 0, 0);
    logsoftmax_kernel<<<cdiv(R, 256), 256, 0, stream>>>(arB, (float*)d_out, R, 13);
  }
}
